// OracleNetwork_44805098832017
// MI455X (gfx1250) — matrix operand order repaired and checked by reading, not compiled, not run on hardware
//
#include <hip/hip_runtime.h>
#include <hip/hip_bf16.h>

// ---------------------------------------------------------------------------
// Particle filter (N=512 particles, T=128 sequential steps).
// Dominant FLOPs: per-step hidden GEMM [512x1024]@[1024x512] done with
// v_wmma_f32_16x16x32_bf16 (bf16 in, f32 accumulate). Weights converted to a
// transposed bf16 copy once per call so WMMA fragments load contiguously.
// ---------------------------------------------------------------------------

typedef __attribute__((ext_vector_type(16))) __bf16 v16bf;
typedef __attribute__((ext_vector_type(8)))  __bf16 v8bf;
typedef __attribute__((ext_vector_type(8)))  float  v8f;

#define NPART 512
#define TSTEP 128
#define HID1  1024
#define HID2  512

__device__ __forceinline__ float sigmoidf_(float x) {
    return 1.f / (1.f + __expf(-x));
}

// open-circuit voltage model
__device__ __forceinline__ float voc_f(float s) {
    const float V_L = 3.4f, V_0 = 4.2f, GAMMA = 0.9f, ALPHA = 0.15f, BETA = 15.0f;
    return V_L + (V_0 - V_L) * __expf(GAMMA * (s - 1.f))
         + ALPHA * V_L * (s - 1.f)
         + (1.f - ALPHA) * V_L * (__expf(-BETA) - __expf(-BETA * sqrtf(s)));
}

// state propagation + clamp to (0,1]
__device__ __forceinline__ float prop_soc(float soc, float Iprev, float Vprev, float nz) {
    float s = soc - Iprev * Vprev * 2e-5f + 0.05f * nz;
    return s > 1.f ? 1.f : (s < 0.f ? 1e-10f : s);
}

// --------------------------------------------------------------------------
// One-time prep: w2 fp32 [K=1024][N=512] row-major -> W2T bf16 [N][K]
// (transposed so a WMMA B-fragment lane reads 16 contiguous bf16 = 32 bytes)
// --------------------------------------------------------------------------
__global__ void k_prep(const float* __restrict__ w2, __bf16* __restrict__ W2T) {
    int o = blockIdx.x * 256 + threadIdx.x;     // 0 .. 512*1024-1
    int n = o >> 10;
    int k = o & 1023;
    W2T[o] = (__bf16)w2[k * HID2 + n];
}

// --------------------------------------------------------------------------
// Layer 1: soc propagation (recomputed per hidden unit, cheap) + h1 = sigmoid
// Writes H1 bf16 [N=512][K=1024] row-major (WMMA A operand).
// t < 0 : init pass (soc = soc_init, no propagation)
// --------------------------------------------------------------------------
__global__ void k_layer1(int t,
                         const float* __restrict__ soc_init,
                         const float* __restrict__ cur,
                         const float* __restrict__ noise,
                         const float* __restrict__ w1,
                         const float* __restrict__ b1,
                         const float* __restrict__ soc_state,
                         const float* __restrict__ V_state,
                         __bf16* __restrict__ H1) {
    int gid = blockIdx.x * 256 + threadIdx.x;   // 512*1024 threads
    int p = gid >> 10;
    int j = gid & 1023;
    float soc;
    if (t < 0) {
        soc = soc_init[p];
    } else {
        float Iprev = cur[t > 0 ? t - 1 : 0];
        soc = prop_soc(soc_state[p], Iprev, V_state[p], noise[t * NPART + p]);
    }
    float sI = cur[t < 0 ? 0 : t];              // (I - 0)/(1 - 0)
    float h = sigmoidf_(soc * w1[j] + sI * w1[HID1 + j] + b1[j]);
    H1[p * HID1 + j] = (__bf16)h;
}

// --------------------------------------------------------------------------
// Hidden GEMM: H2 = sigmoid(H1 @ W2 + b2),  M=512, N=512, K=1024.
// One 16x16 output tile per wave32; 32 x v_wmma_f32_16x16x32_bf16 over K.
// A fragment: lane half h loads K chunk  [h*16, h*16+16) of H1 row m.
// B fragment: lane half h loads K chunk  [h*16, h*16+16) of W2T row n.
// Output stored transposed (H2T[n][m]) so the tail kernel's dot coalesces.
// --------------------------------------------------------------------------
__global__ void k_gemm2(const __bf16* __restrict__ H1,
                        const __bf16* __restrict__ W2T,
                        const float*  __restrict__ b2,
                        float*        __restrict__ H2T) {
    int lane = threadIdx.x & 31;
    int wave = threadIdx.x >> 5;
    int tile = blockIdx.x * 8 + wave;           // 0 .. 1023
    int tm = tile >> 5;                         // 0 .. 31 (M tiles)
    int tn = tile & 31;                         // 0 .. 31 (N tiles)
    int hf = lane >> 4;                         // K-half selector
    int l  = lane & 15;

    const __bf16* arow = H1  + (tm * 16 + l) * HID1 + hf * 16;
    const __bf16* brow = W2T + (tn * 16 + l) * HID1 + hf * 16;

    v8f c = {};
#pragma unroll 4
    for (int k0 = 0; k0 < HID1; k0 += 32) {
        v8bf alo = *(const v8bf*)(arow + k0);
        v8bf ahi = *(const v8bf*)(arow + k0 + 8);
        v16bf a = __builtin_shufflevector(alo, ahi,
                    0, 1, 2, 3, 4, 5, 6, 7, 8, 9, 10, 11, 12, 13, 14, 15);
        v16bf b = *(const v16bf*)(brow + k0);
        c = __builtin_amdgcn_wmma_f32_16x16x32_bf16(
                false, a, false, b, (short)0, c, false, false);
    }

    int ncol = tn * 16 + l;
    float bias = b2[ncol];
#pragma unroll
    for (int r = 0; r < 8; ++r) {
        int m = tm * 16 + r + hf * 8;           // D VGPR r -> row r (+8 for hi half)
        H2T[ncol * NPART + m] = sigmoidf_(c[r] + bias);
    }
}

// --------------------------------------------------------------------------
// Tail: Z = h2@w3 + b3, V = voc - I*Z, logsumexp loss, softmax, cumsum,
// systematic resampling, state/history update. Single 512-thread block.
// t < 0 : init pass (store V0, soc_state = soc_init, no loss/resample)
// --------------------------------------------------------------------------
__global__ void __launch_bounds__(512)
k_tail(int t,
       const float* __restrict__ soc_init,
       const float* __restrict__ cur,
       const float* __restrict__ vmeas,
       const float* __restrict__ noise,
       const float* __restrict__ u,
       const float* __restrict__ w3,
       const float* __restrict__ b3,
       const float* __restrict__ H2T,
       float* __restrict__ soc_state,
       float* __restrict__ V_state,
       float* __restrict__ losses,
       float* __restrict__ out_volt,
       float* __restrict__ out_soc) {
    __shared__ float sh[NPART];
    __shared__ float cum[NPART];
    __shared__ float socs[NPART];
    int p = threadIdx.x;

    // output layer: Z[p] = sum_j H2T[j][p] * w3[j]  (coalesced over p)
    float z = b3[0];
#pragma unroll 8
    for (int j = 0; j < HID2; ++j)
        z += H2T[j * NPART + p] * w3[j];

    float soc;
    if (t < 0) {
        soc = soc_init[p];
    } else {
        float Iprev = cur[t > 0 ? t - 1 : 0];
        soc = prop_soc(soc_state[p], Iprev, V_state[p], noise[t * NPART + p]);
    }
    float It = cur[t < 0 ? 0 : t];
    float V = voc_f(soc) - It * z;

    if (t < 0) {                 // init pass: just seed the carry
        V_state[p]   = V;
        soc_state[p] = soc;
        return;
    }

    // log weights: log(NU) - 0.5*((V - vmeas)/G_STD)^2,  NU = 1/(0.02*sqrt(2pi))
    float d = (V - vmeas[t]) * 50.f;            // / 0.02
    float logW = 2.9930854f - 0.5f * d * d;     // ln(19.94711402)

    // logsumexp: max then sum
    sh[p] = logW; __syncthreads();
    for (int s = 256; s > 0; s >>= 1) {
        if (p < s) sh[p] = fmaxf(sh[p], sh[p + s]);
        __syncthreads();
    }
    float mx = sh[0]; __syncthreads();
    float e = __expf(logW - mx);
    sh[p] = e; __syncthreads();
    for (int s = 256; s > 0; s >>= 1) {
        if (p < s) sh[p] += sh[p + s];
        __syncthreads();
    }
    float ssum = sh[0]; __syncthreads();
    if (p == 0) losses[t] = mx + __logf(ssum) - 6.2383246f;   // - ln(512)

    // softmax + inclusive cumsum (Hillis-Steele)
    float wn = e / ssum;
    cum[p] = wn;
    socs[p] = soc;
    __syncthreads();
    for (int off = 1; off < NPART; off <<= 1) {
        float a = cum[p];
        float b = (p >= off) ? cum[p - off] : 0.f;
        __syncthreads();
        cum[p] = a + b;
        __syncthreads();
    }

    // systematic resampling: lower_bound(cum, (p + u_t)/N), clipped
    float pos = ((float)p + u[t]) * (1.f / (float)NPART);
    int lo = 0, hi = NPART;
    while (lo < hi) {
        int mid = (lo + hi) >> 1;
        if (cum[mid] < pos) lo = mid + 1; else hi = mid;
    }
    int idx = lo > NPART - 1 ? NPART - 1 : lo;
    float soc_r = socs[idx];

    out_volt[p * TSTEP + t] = V;
    out_soc[p * TSTEP + t]  = soc_r;
    V_state[p]   = V;
    soc_state[p] = soc_r;
}

__global__ void k_loss(const float* __restrict__ losses, float* __restrict__ out) {
    if (threadIdx.x == 0) {
        float s = 0.f;
        for (int i = 0; i < TSTEP; ++i) s += losses[i];
        out[0] = s;
    }
}

// --------------------------------------------------------------------------
extern "C" void kernel_launch(void* const* d_in, const int* in_sizes, int n_in,
                              void* d_out, int out_size, void* d_ws, size_t ws_size,
                              hipStream_t stream) {
    (void)in_sizes; (void)n_in; (void)out_size; (void)ws_size;
    const float* soc_init = (const float*)d_in[0];   // [512]
    const float* cur      = (const float*)d_in[1];   // [128]
    const float* vmeas    = (const float*)d_in[2];   // [128]
    const float* noise    = (const float*)d_in[3];   // [128*512]
    const float* uarr     = (const float*)d_in[4];   // [128]
    const float* w1       = (const float*)d_in[5];   // [2*1024]
    const float* b1       = (const float*)d_in[6];   // [1024]
    const float* w2       = (const float*)d_in[7];   // [1024*512]
    const float* b2       = (const float*)d_in[8];   // [512]
    const float* w3       = (const float*)d_in[9];   // [512]
    const float* b3       = (const float*)d_in[10];  // [1]

    char* ws = (char*)d_ws;
    __bf16* W2T = (__bf16*)(ws);                         // 1 MB
    __bf16* H1  = (__bf16*)(ws + (1u << 20));            // 1 MB
    float*  H2T = (float*)(ws + (2u << 20));             // 1 MB
    float*  soc_state = (float*)(ws + 3u * (1u << 20));  // 2 KB
    float*  V_state   = soc_state + NPART;               // 2 KB
    float*  losses    = V_state + NPART;                 // 512 B

    float* out      = (float*)d_out;
    float* out_volt = out + 1;
    float* out_soc  = out + 1 + NPART * TSTEP;

    // one-time weight transpose/convert
    k_prep<<<2048, 256, 0, stream>>>(w2, W2T);

    // init pass (t = -1) then T sequential steps
    for (int t = -1; t < TSTEP; ++t) {
        k_layer1<<<2048, 256, 0, stream>>>(t, soc_init, cur, noise, w1, b1,
                                           soc_state, V_state, H1);
        k_gemm2<<<128, 256, 0, stream>>>(H1, W2T, b2, H2T);
        k_tail<<<1, 512, 0, stream>>>(t, soc_init, cur, vmeas, noise, uarr,
                                      w3, b3, H2T, soc_state, V_state, losses,
                                      out_volt, out_soc);
    }
    k_loss<<<1, 32, 0, stream>>>(losses, out);
}
